// RelInteractionDebug4_27032524161363
// MI455X (gfx1250) — matrix operand order repaired and checked by reading, not compiled, not run on hardware
//
#include <hip/hip_runtime.h>
#include <hip/hip_bf16.h>
#include <stdint.h>

// ---------------------------------------------------------------------------
// CDNA5 (gfx1250): all heavy GEMMs route through one tiled bf16 WMMA kernel
// (v_wmma_f32_16x16x32_bf16, fp32 accumulate), double-buffered LDS, b128
// global staging on full tiles.  Requires K % 32 == 0 (true for every GEMM
// in this net).  MREP stacks 16-row C tiles per wave: TM = 32*MREP, so the
// big-M GEMMs stream their (L2-overflowing) weight panels 4x less often.
// All fragment gathers are hoisted ahead of the WMMA burst so the 4 matrix
// ops issue back-to-back behind a single s_wait_dscnt.
// ---------------------------------------------------------------------------

typedef __attribute__((ext_vector_type(16))) __bf16 v16bf;
typedef __attribute__((ext_vector_type(8)))  float  v8f;

#define TN 64
#define TK 32
#define ASTR 36   // padded LDS row strides (bf16 units)
#define BSTR 36

union PK2 { __bf16 h[2]; unsigned u; };
union PK4 { __bf16 h[4]; uint2 u2; };

template<int MREP, bool ACONTIG>
__global__ __launch_bounds__(256) void gemm_bf16_wmma(
    const float* __restrict__ A, long long sam, long long sak, long long abs_,
    const float* __restrict__ B, long long sbk, long long sbn, long long bbs,
    float* __restrict__ C, long long ldc, long long cbs,
    const float* __restrict__ bias, int bias_mode, int relu,
    int M, int N, int K)
{
    constexpr int TMv = 32 * MREP;
    __shared__ __bf16 As[2][TMv * ASTR];   // [m][k]
    __shared__ __bf16 Bs[2][TN * BSTR];    // [n][k] (transposed: k contiguous)

    const int bz = blockIdx.z;
    A += (long long)bz * abs_;
    B += (long long)bz * bbs;
    C += (long long)bz * cbs;

    const int tid  = threadIdx.x;
    const int wid  = tid >> 5;
    const int lane = tid & 31;
    const int wm   = wid & 1;            // 2 wave-rows * 16 = 32-row slab
    const int wn   = wid >> 1;           // 4 wave-cols * 16 = TN
    const int m0   = blockIdx.y * TMv;
    const int n0   = blockIdx.x * TN;
    const int hi   = lane >> 4;
    const int lid  = lane & 15;

    const bool full = (m0 + TMv <= M) && (n0 + TN <= N);

    float ar[MREP][4];   // register staging for next A tile
    float br[8];         // register staging for next B tile

    auto loadA = [&](int k0) {
        if (ACONTIG && full) {
            int row = tid >> 3, kk = (tid & 7) * 4;          // sak == 1
#pragma unroll
            for (int j = 0; j < MREP; ++j) {
                const float4 v = *(const float4*)
                    &A[(long long)(m0 + 32 * j + row) * sam + (k0 + kk)];
                ar[j][0] = v.x; ar[j][1] = v.y; ar[j][2] = v.z; ar[j][3] = v.w;
            }
            return;
        }
#pragma unroll
        for (int j = 0; j < MREP; ++j)
#pragma unroll
            for (int q = 0; q < 4; ++q) {                    // branchless, clamped
                int i = tid + q * 256;
                int row = 32 * j + (i >> 5), kk = i & 31;
                int m = m0 + row;
                int mc = m < M ? m : (M - 1);
                float v = A[(long long)mc * sam + (long long)(k0 + kk) * sak];
                ar[j][q] = (m < M) ? v : 0.f;
            }
    };
    auto storeA = [&](int buf) {
        if (ACONTIG && full) {
            int row = tid >> 3, kk = (tid & 7) * 4;
#pragma unroll
            for (int j = 0; j < MREP; ++j) {
                PK4 p;
                p.h[0] = (__bf16)ar[j][0]; p.h[1] = (__bf16)ar[j][1];
                p.h[2] = (__bf16)ar[j][2]; p.h[3] = (__bf16)ar[j][3];
                *(uint2*)&As[buf][(32 * j + row) * ASTR + kk] = p.u2;   // 8B aligned
            }
            return;
        }
#pragma unroll
        for (int j = 0; j < MREP; ++j)
#pragma unroll
            for (int q = 0; q < 4; ++q) {
                int i = tid + q * 256;
                As[buf][(32 * j + (i >> 5)) * ASTR + (i & 31)] = (__bf16)ar[j][q];
            }
    };
    auto loadB = [&](int k0) {
        if (full) {                                          // sbn == 1 here
            int nn = (tid & 15) * 4, kk = (tid >> 4) * 2;
            const float* p = &B[(long long)(k0 + kk) * sbk + (n0 + nn)];
            const float4 v0 = *(const float4*)p;
            const float4 v1 = *(const float4*)(p + sbk);
            br[0] = v0.x; br[1] = v0.y; br[2] = v0.z; br[3] = v0.w;
            br[4] = v1.x; br[5] = v1.y; br[6] = v1.z; br[7] = v1.w;
            return;
        }
#pragma unroll
        for (int j = 0; j < 8; ++j) {
            int i = tid + j * 256;
            int nn = i & 63, kk = i >> 6;
            int n = n0 + nn;
            int nc = n < N ? n : (N - 1);
            float v = B[(long long)(k0 + kk) * sbk + (long long)nc * sbn];
            br[j] = (n < N) ? v : 0.f;
        }
    };
    auto storeB = [&](int buf) {
        if (full) {
            int nn = (tid & 15) * 4, kk = (tid >> 4) * 2;
#pragma unroll
            for (int j = 0; j < 4; ++j) {                    // pack (k,k+1) pairs
                PK2 p;
                p.h[0] = (__bf16)br[j];
                p.h[1] = (__bf16)br[4 + j];
                *(unsigned*)&Bs[buf][(nn + j) * BSTR + kk] = p.u;   // 4B aligned
            }
            return;
        }
#pragma unroll
        for (int j = 0; j < 8; ++j) {
            int i = tid + j * 256;
            Bs[buf][(i & 63) * BSTR + (i >> 6)] = (__bf16)br[j];
        }
    };

    v8f acc[MREP];
#pragma unroll
    for (int j = 0; j < MREP; ++j) acc[j] = (v8f){0.f,0.f,0.f,0.f,0.f,0.f,0.f,0.f};

    const int nK = K / TK;       // K % 32 == 0 guaranteed by caller
    loadA(0); loadB(0); storeA(0); storeB(0);
    __syncthreads();

    for (int kt = 0; kt < nK; ++kt) {
        const int cur = kt & 1, nxt = cur ^ 1;
        const bool more = (kt + 1) < nK;
        if (more) {                       // issue next tile's global loads early
            loadA((kt + 1) * TK);
            loadB((kt + 1) * TK);
            if (lane == 0 && (kt + 2) < nK)
                __builtin_prefetch(&B[(long long)(kt + 2) * TK * sbk + (long long)n0 * sbn], 0, 1);
        }

        // ---- hoist ALL fragment gathers (contiguous k = 16*hi + e in A and B) -------
        const __bf16* bp = &Bs[cur][(wn * 16 + lid) * BSTR + hi * 16];
        v16bf bfrag;
#pragma unroll
        for (int e = 0; e < 16; ++e) bfrag[e] = bp[e];

        v16bf afrag[MREP];
#pragma unroll
        for (int j = 0; j < MREP; ++j) {
            const __bf16* ap = &As[cur][(32 * j + wm * 16 + lid) * ASTR];
#pragma unroll
            for (int q = 0; q < 8; ++q) {
                int kp = 2 * q + hi * 16;
                afrag[j][2 * q]     = ap[kp];
                afrag[j][2 * q + 1] = ap[kp + 1];
            }
        }

        // ---- WMMA burst: back-to-back matrix ops --------------------------
#pragma unroll
        for (int j = 0; j < MREP; ++j)
            acc[j] = __builtin_amdgcn_wmma_f32_16x16x32_bf16(
                false, afrag[j], false, bfrag, (short)0, acc[j], false, false);

        if (more) {                       // convert + store into other buffer
            storeA(nxt);
            storeB(nxt);
        }
        __syncthreads();
    }

    // ---- epilogue: C/D layout  n = 16*wn + lid,  m = 16*wm + 8*hi + r -----
    int n_g = n0 + wn * 16 + lid;
    if (n_g < N) {
#pragma unroll
        for (int j = 0; j < MREP; ++j)
#pragma unroll
            for (int r = 0; r < 8; ++r) {
                int m_g = m0 + 32 * j + wm * 16 + hi * 8 + r;
                if (m_g < M) {
                    float v = acc[j][r];
                    if (bias_mode == 1)      v += bias[n_g];
                    else if (bias_mode == 2) v += bias[m_g];
                    if (relu) v = fmaxf(v, 0.f);
                    C[(long long)m_g * ldc + n_g] = v;
                }
            }
    }
}

// ---- dynamic depthwise 3x3 conv over gathered reduced fmaps ---------------
__global__ __launch_bounds__(256) void k_dwconv(
    const float* __restrict__ red, const float* __restrict__ dwf,
    const float* __restrict__ rois, float* __restrict__ y)
{
    int nc = blockIdx.x;                 // n*512 + c
    int n = nc >> 9, c = nc & 511;
    int im = (int)rois[n * 5];           // im_inds
    const float* w9 = dwf + (long long)nc * 9;
    float w[9];
#pragma unroll
    for (int t = 0; t < 9; ++t) w[t] = w9[t];
    const float* src = red + ((long long)im * 512 + c) * 1444;
    float* dst = y + (long long)nc * 1444;
    for (int p = threadIdx.x; p < 1444; p += 256) {
        int h = p / 38, wq = p - h * 38;
        float acc = 0.f;
#pragma unroll
        for (int ky = 0; ky < 3; ++ky) {
            int hh = h + ky - 1;
            if (hh < 0 || hh >= 38) continue;
#pragma unroll
            for (int kx = 0; kx < 3; ++kx) {
                int ww = wq + kx - 1;
                if (ww < 0 || ww >= 38) continue;
                acc += src[hh * 38 + ww] * w[ky * 3 + kx];
            }
        }
        dst[p] = acc;
    }
}

// ---- BatchNorm2d (training stats over N,H,W): deterministic reductions ----
__global__ __launch_bounds__(256) void k_bnstats(
    const float* __restrict__ x, float* __restrict__ sum, float* __restrict__ sq)
{
    __shared__ float s1[256], s2[256];
    int c = blockIdx.x;
    float a = 0.f, b = 0.f;
    for (int i = threadIdx.x; i < 32 * 1444; i += 256) {
        int n = i / 1444, p = i - n * 1444;
        float v = x[((long long)n * 512 + c) * 1444 + p];
        a += v; b += v * v;
    }
    s1[threadIdx.x] = a; s2[threadIdx.x] = b;
    __syncthreads();
    for (int off = 128; off > 0; off >>= 1) {
        if (threadIdx.x < off) {
            s1[threadIdx.x] += s1[threadIdx.x + off];
            s2[threadIdx.x] += s2[threadIdx.x + off];
        }
        __syncthreads();
    }
    if (threadIdx.x == 0) { sum[c] = s1[0]; sq[c] = s2[0]; }
}

__global__ __launch_bounds__(256) void k_bnapply(
    float* __restrict__ x, const float* __restrict__ sum, const float* __restrict__ sq,
    const float* __restrict__ g, const float* __restrict__ b, int relu)
{
    long long idx = (long long)blockIdx.x * 256 + threadIdx.x;
    const long long total = 32LL * 512 * 1444;
    if (idx >= total) return;
    int c = (int)((idx / 1444) & 511);
    const float inv = 1.f / (32.f * 1444.f);
    float m = sum[c] * inv;
    float v = sq[c] * inv - m * m;
    float o = (x[idx] - m) * rsqrtf(v + 1e-5f) * g[c] + b[c];
    if (relu) o = fmaxf(o, 0.f);
    x[idx] = o;
}

// ---- antialiased linear 38->7 resize + sub/obj gather into so[r][c][hw] ---
__global__ __launch_bounds__(256) void k_resize_so(
    const float* __restrict__ z, const int* __restrict__ rel_inds,
    float* __restrict__ so)
{
    long long idx = (long long)blockIdx.x * 256 + threadIdx.x;
    const long long total = 256LL * 1024 * 49;
    if (idx >= total) return;
    int p = (int)(idx % 49);
    int c = (int)((idx / 49) % 1024);
    int r = (int)(idx / (49LL * 1024));
    int oh = p / 7, ow = p - oh * 7;
    int obj = (c < 512) ? rel_inds[r * 3 + 1] : rel_inds[r * 3 + 2];
    int ch = c & 511;
    const float* src = z + ((long long)obj * 512 + ch) * 1444;
    const float scale = 38.0f / 7.0f;

    float wy[16], wx[16];
    float yc = (oh + 0.5f) * scale - 0.5f;
    int iy0 = (int)ceilf(yc - scale), iy1 = (int)floorf(yc + scale);
    int ny = 0; float wys = 0.f;
    for (int i = iy0; i <= iy1 && ny < 16; ++i, ++ny) {
        float w = 1.f - fabsf((float)i - yc) / scale;
        wy[ny] = (w > 0.f) ? w : 0.f; wys += wy[ny];
    }
    float xc = (ow + 0.5f) * scale - 0.5f;
    int ix0 = (int)ceilf(xc - scale), ix1 = (int)floorf(xc + scale);
    int nx = 0; float wxs = 0.f;
    for (int i = ix0; i <= ix1 && nx < 16; ++i, ++nx) {
        float w = 1.f - fabsf((float)i - xc) / scale;
        wx[nx] = (w > 0.f) ? w : 0.f; wxs += wx[nx];
    }
    float acc = 0.f;
    for (int a = 0; a < ny; ++a) {
        int ih = iy0 + a; ih = ih < 0 ? 0 : (ih > 37 ? 37 : ih);
        float row = 0.f;
        for (int b = 0; b < nx; ++b) {
            int iw = ix0 + b; iw = iw < 0 ? 0 : (iw > 37 ? 37 : iw);
            row += wx[b] * src[ih * 38 + iw];
        }
        acc += wy[a] * row;
    }
    so[idx] = acc / (wys * wxs);
}

// ---- concat(sub, obj) gather for mapping_x --------------------------------
__global__ __launch_bounds__(256) void k_catso(
    const float* __restrict__ subobj, const int* __restrict__ rel_inds,
    float* __restrict__ catso)
{
    int idx = blockIdx.x * 256 + threadIdx.x;
    if (idx >= 256 * 1024) return;
    int r = idx >> 10, j = idx & 1023;
    int obj = (j < 512) ? rel_inds[r * 3 + 1] : rel_inds[r * 3 + 2];
    catso[idx] = subobj[obj * 1024 + j];   // cols 0..511 = sub half, 512..1023 = obj half
}

// ---- triple = relu(a+b) - (a-b)^2 ----------------------------------------
__global__ __launch_bounds__(256) void k_triple(
    const float* __restrict__ loi, const float* __restrict__ lri, float* __restrict__ tri)
{
    int idx = blockIdx.x * 256 + threadIdx.x;
    if (idx >= 256 * 1536) return;
    float a = loi[idx], b = lri[idx];
    float d = a - b;
    tri[idx] = fmaxf(a + b, 0.f) - d * d;
}

// ---------------------------------------------------------------------------
extern "C" void kernel_launch(void* const* d_in, const int* in_sizes, int n_in,
                              void* d_out, int out_size, void* d_ws, size_t ws_size,
                              hipStream_t stream)
{
    const float* fmaps  = (const float*)d_in[0];
    const float* rois   = (const float*)d_in[1];
    const int*   rel    = (const int*)  d_in[2];
    const float* rof    = (const float*)d_in[3];
    const float* lof    = (const float*)d_in[4];
    const float* W_dw   = (const float*)d_in[5];  const float* b_dw  = (const float*)d_in[6];
    const float* W_pw   = (const float*)d_in[7];  const float* b_pw  = (const float*)d_in[8];
    const float* W_red  = (const float*)d_in[9];  const float* b_red = (const float*)d_in[10];
    const float* g_dw   = (const float*)d_in[11]; const float* be_dw = (const float*)d_in[12];
    const float* g_pw   = (const float*)d_in[13]; const float* be_pw = (const float*)d_in[14];
    const float* W_rec  = (const float*)d_in[15]; const float* b_rec = (const float*)d_in[16];
    const float* W_fc1  = (const float*)d_in[17]; const float* b_fc1 = (const float*)d_in[18];
    const float* W_fc2  = (const float*)d_in[19]; const float* b_fc2 = (const float*)d_in[20];
    const float* W_post = (const float*)d_in[21]; const float* b_post= (const float*)d_in[22];
    const float* W_map  = (const float*)d_in[23]; const float* b_map = (const float*)d_in[24];
    const float* W_rr   = (const float*)d_in[25]; const float* b_rr  = (const float*)d_in[26];
    const float* W_comp = (const float*)d_in[27]; const float* b_comp= (const float*)d_in[28];

    float* ws = (float*)d_ws;
    const long long HW = 1444;
    float* dwf    = ws;                              // 32*4608
    float* pwf    = dwf  + 32LL * 4608;              // 32*262144  [n][o][c]
    float* red    = pwf  + 32LL * 262144;            // 2*512*1444
    float* ybuf   = red  + 2LL * 512 * HW;           // 32*512*1444
    float* zbuf   = ybuf + 32LL * 512 * HW;          // 32*512*1444
    float* sobuf  = ybuf;                            // alias (y dead): 256*1024*49
    float* flat   = ybuf + 256LL * 1024 * 49;        // alias: 256*25088
    float* smalls = zbuf + 32LL * 512 * HW;
    float* bsum   = smalls;                          // 512
    float* bsq    = bsum + 512;                      // 512
    float* subobj = bsq + 512;                       // 32*1024
    float* catso  = subobj + 32LL * 1024;            // 256*1024
    float* h1     = catso  + 256LL * 1024;           // 256*4096
    float* relin  = h1     + 256LL * 4096;           // 256*4096
    float* loi    = relin  + 256LL * 4096;           // 256*1536
    float* lri    = loi    + 256LL * 1536;           // 256*1536
    float* tri    = lri    + 256LL * 1536;           // 256*1536

    auto gemm = [&](int mrep, bool acontig,
                    const float* A, long long sam, long long sak, long long abs_,
                    const float* B, long long sbk, long long sbn, long long bbs,
                    float* C, long long ldc, long long cbs,
                    const float* bias, int bmode, int relu,
                    int M, int N, int K, int batch) {
        int tm = 32 * mrep;
        dim3 g((N + TN - 1) / TN, (M + tm - 1) / tm, batch);
        if (mrep == 4 && acontig)
            gemm_bf16_wmma<4, true><<<g, dim3(256), 0, stream>>>(
                A, sam, sak, abs_, B, sbk, sbn, bbs, C, ldc, cbs, bias, bmode, relu, M, N, K);
        else if (mrep == 4)
            gemm_bf16_wmma<4, false><<<g, dim3(256), 0, stream>>>(
                A, sam, sak, abs_, B, sbk, sbn, bbs, C, ldc, cbs, bias, bmode, relu, M, N, K);
        else
            gemm_bf16_wmma<1, true><<<g, dim3(256), 0, stream>>>(
                A, sam, sak, abs_, B, sbk, sbn, bbs, C, ldc, cbs, bias, bmode, relu, M, N, K);
    };

    // dw = rof @ W_dw + b_dw                         [32,4608]
    gemm(1, true, rof, 512, 1, 0, W_dw, 4608, 1, 0, dwf, 4608, 0, b_dw, 1, 0, 32, 4608, 512, 1);
    // pw = rof @ W_pw + b_pw                         [32,262144] == [n][o][c]
    gemm(1, true, rof, 512, 1, 0, W_pw, 262144, 1, 0, pwf, 262144, 0, b_pw, 1, 0, 32, 262144, 512, 1);
    // red[i][o][p] = relu(sum_c W_red[c][o]*fmaps[i][c][p] + b_red[o])
    gemm(4, false, W_red, 1, 512, 0, fmaps, HW, 1, 512 * HW, red, HW, 512 * HW,
         b_red, 2, 1, 512, 1444, 512, 2);
    // dynamic depthwise 3x3 over red[im_inds]
    k_dwconv<<<dim3(32 * 512), dim3(256), 0, stream>>>(red, dwf, rois, ybuf);
    // BN(y)
    k_bnstats<<<dim3(512), dim3(256), 0, stream>>>(ybuf, bsum, bsq);
    k_bnapply<<<dim3((32 * 512 * 1444 + 255) / 256), dim3(256), 0, stream>>>(
        ybuf, bsum, bsq, g_dw, be_dw, 0);
    // z[n][o][p] = sum_c pw[n][o][c] * y[n][c][p]    (batched per object)
    gemm(4, true, pwf, 512, 1, 262144, ybuf, HW, 1, 512 * HW, zbuf, HW, 512 * HW,
         nullptr, 0, 0, 512, 1444, 512, 32);
    // relu(BN(z))
    k_bnstats<<<dim3(512), dim3(256), 0, stream>>>(zbuf, bsum, bsq);
    k_bnapply<<<dim3((32 * 512 * 1444 + 255) / 256), dim3(256), 0, stream>>>(
        zbuf, bsum, bsq, g_pw, be_pw, 1);
    // so[r][c][hw] : antialiased 38->7 resize of z[sub]/z[obj]
    k_resize_so<<<dim3((int)((256LL * 1024 * 49 + 255) / 256)), dim3(256), 0, stream>>>(
        zbuf, rel, sobuf);
    // flat[r][o*49+hw] = sum_c W_rec[c][o]*so[r][c][hw] + b_rec[o]  (batched per rel)
    gemm(4, false, W_rec, 1, 512, 0, sobuf, 49, 1, 1024LL * 49, flat, 49, 25088,
         b_rec, 2, 0, 512, 49, 1024, 256);
    // h1 = relu(flat @ W_fc1 + b_fc1)
    gemm(4, true, flat, 25088, 1, 0, W_fc1, 4096, 1, 0, h1, 4096, 0,
         b_fc1, 1, 1, 256, 4096, 25088, 1);
    // rel_input = h1 @ W_fc2 + b_fc2
    gemm(4, true, h1, 4096, 1, 0, W_fc2, 4096, 1, 0, relin, 4096, 0,
         b_fc2, 1, 0, 256, 4096, 4096, 1);
    // last_rel_input = rel_input @ W_rr + b_rr
    gemm(4, true, relin, 4096, 1, 0, W_rr, 1536, 1, 0, lri, 1536, 0,
         b_rr, 1, 0, 256, 1536, 4096, 1);
    // subobj = lof @ W_post + b_post
    gemm(1, true, lof, 512, 1, 0, W_post, 1024, 1, 0, subobj, 1024, 0,
         b_post, 1, 0, 32, 1024, 512, 1);
    // concat(sub, obj) per relation
    k_catso<<<dim3(1024), dim3(256), 0, stream>>>(subobj, rel, catso);
    // last_obj_input = catso @ W_map + b_map
    gemm(4, true, catso, 1024, 1, 0, W_map, 1536, 1, 0, loi, 1536, 0,
         b_map, 1, 0, 256, 1536, 1024, 1);
    // triple = relu(a+b) - (a-b)^2
    k_triple<<<dim3((256 * 1536 + 255) / 256), dim3(256), 0, stream>>>(loi, lri, tri);
    // out = triple @ W_comp + b_comp                 [256,51] -> d_out
    gemm(4, true, tri, 1536, 1, 0, W_comp, 51, 1, 0, (float*)d_out, 51, 0,
         b_comp, 1, 0, 256, 51, 1536, 1);
}
